// Model_10307921510433
// MI455X (gfx1250) — matrix operand order repaired and checked by reading, not compiled, not run on hardware
//
#include <hip/hip_runtime.h>

#define VOCAB 32000
#define CDIM 384
#define LNUM 6
#define BDIM 32
#define TDIM 256
#define ROWS (BDIM * TDIM)   // 8192

typedef __attribute__((ext_vector_type(16))) __bf16 v16bf;
typedef __attribute__((ext_vector_type(8)))  float  v8f;

#define WMMA_BF16(c, a, b) \
  __builtin_amdgcn_wmma_f32_16x16x32_bf16(false, (a), false, (b), (short)0, (c), false, false)

// ---- CDNA5 WMMA fragment loaders (wave32; contiguous k = 16*h + s in both A and B) ----

// A (bf16, 16x32) from row-major global src: per lane two 16B-contiguous chunks
__device__ __forceinline__ v16bf load_a_bf16(const __bf16* __restrict__ src, int ld,
                                             int m0, int k0, int lane) {
  int m = lane & 15, hh = lane >> 4;
  const __bf16* row = src + (size_t)(m0 + m) * ld + k0;
  v16bf a;
#pragma unroll
  for (int j = 0; j < 8; ++j) a[j] = row[hh * 16 + j];           // K = hh*16 + 0..7
#pragma unroll
  for (int j = 0; j < 8; ++j) a[8 + j] = row[hh * 16 + 8 + j];  // K = hh*16 + 8..15
  return a;
}

// A (bf16, 16x32) from LDS bf16 shadow of h
__device__ __forceinline__ v16bf load_a_bf16lds(const __bf16* __restrict__ src, int ld,
                                                int k0, int lane) {
  int m = lane & 15, hh = lane >> 4;
  const __bf16* row = src + m * ld + k0;
  v16bf a;
#pragma unroll
  for (int j = 0; j < 8; ++j) a[j] = row[hh * 16 + j];
#pragma unroll
  for (int j = 0; j < 8; ++j) a[8 + j] = row[hh * 16 + 8 + j];
  return a;
}

// B (bf16, 32x16) from pre-packed fragment tiles: 32 contiguous bytes per lane
__device__ __forceinline__ v16bf load_b_packed(const __bf16* __restrict__ packed,
                                               int ntiles, int kt, int nt, int lane) {
  const __bf16* p = packed + (((size_t)kt * ntiles + nt) * 32 + lane) * 16;
  v16bf b;
#pragma unroll
  for (int j = 0; j < 16; ++j) b[j] = p[j];
  return b;
}

// ------------------------- prep kernels -------------------------

// Fused cast + pack of a row-major [K][N] fp32 matrix into WMMA-B fragment tiles.
__global__ __launch_bounds__(256) void pack_b(const float* __restrict__ W,
                                              __bf16* __restrict__ out, int K, int N) {
  size_t per = (size_t)K * N;
  const float* Wl = W + (size_t)blockIdx.y * per;
  __bf16* ol = out + (size_t)blockIdx.y * per;
  int i = blockIdx.x * 256 + threadIdx.x;
  if (i >= (int)per) return;
  int j = i & 15, lane = (i >> 4) & 31, tile = i >> 9;
  int ntiles = N >> 4;
  int nt = tile % ntiles, kt = tile / ntiles;
  int n = lane & 15, g = lane >> 4;
  int k = kt * 32 + g * 16 + j;
  ol[i] = (__bf16)Wl[(size_t)k * N + nt * 16 + n];
}

// Pack B = E^T from row-major E [N=VOCAB][K=C]: B[k][n] = E[n][k]
__global__ __launch_bounds__(256) void pack_b_T(const float* __restrict__ Erows,
                                                __bf16* __restrict__ out, int K, int N) {
  size_t per = (size_t)K * N;
  size_t i = (size_t)blockIdx.x * 256 + threadIdx.x;
  if (i >= per) return;
  int j = (int)(i & 15);
  int lane = (int)((i >> 4) & 31);
  size_t tile = i >> 9;
  int ntiles = N >> 4;
  int nt = (int)(tile % ntiles), kt = (int)(tile / ntiles);
  int n = lane & 15, g = lane >> 4;
  int k = kt * 32 + g * 16 + j;
  out[i] = (__bf16)Erows[(size_t)(nt * 16 + n) * K + k];
}

// emb_bf16[(t*B+b)*C + c] = (bf16)E[idx[b*T+t]*C + c]
__global__ __launch_bounds__(256) void gather_emb(const int* __restrict__ idx,
                                                  const float* __restrict__ E,
                                                  __bf16* __restrict__ emb) {
  int i = blockIdx.x * 256 + threadIdx.x;
  if (i >= ROWS * CDIM) return;
  int c = i % CDIM;
  int r = i / CDIM;
  int b = r % BDIM;
  int t = r / BDIM;
  emb[i] = (__bf16)E[(size_t)idx[b * TDIM + t] * CDIM + c];
}

// ------------------------- phase-1 GEMMs -------------------------
// do_ln==0: zpre = emb @ Wp_top[l] + bp[l], stored in D-fragment layout
// do_ln==1: scale = LN(emb @ Ws[l] + bs[l]; gs, betas), row-major [t][l][b][c]
__global__ __launch_bounds__(256) void gemm_pre(const __bf16* __restrict__ Aemb,
                                                const __bf16* __restrict__ Bpacked,
                                                const float* __restrict__ bias,
                                                const float* __restrict__ gamma,
                                                const float* __restrict__ beta,
                                                __bf16* __restrict__ out,
                                                int bstride, int do_ln) {
  const int l   = blockIdx.y;
  const int rb  = blockIdx.x;            // 0..511
  const int tid = threadIdx.x;
  const int wave = tid >> 5, lane = tid & 31;
  const int lanen = lane & 15, laneh = lane >> 4;
  const int m0 = rb * 16;
  const __bf16* Bmat = Bpacked + (size_t)l * bstride;

  __shared__ float zbuf[16][CDIM];
  __shared__ float red1[16][16], red2[16][16];

  v8f c0 = {}, c1 = {}, c2 = {};
  const int n0 = wave * 48;
  const int nt0 = wave * 3;
  for (int kc = 0; kc < CDIM; kc += 32) {
    int kt = kc >> 5;
    v16bf a  = load_a_bf16(Aemb, CDIM, m0, kc, lane);
    v16bf b0 = load_b_packed(Bmat, 24, kt, nt0, lane);
    c0 = WMMA_BF16(c0, a, b0);
    v16bf b1 = load_b_packed(Bmat, 24, kt, nt0 + 1, lane);
    c1 = WMMA_BF16(c1, a, b1);
    v16bf b2 = load_b_packed(Bmat, 24, kt, nt0 + 2, lane);
    c2 = WMMA_BF16(c2, a, b2);
  }
  const float* bi = bias + l * CDIM;
  const int t = rb >> 1, bhalf = rb & 1;

  if (!do_ln) {
    float bv0 = bi[n0 + lanen], bv1 = bi[n0 + 16 + lanen], bv2 = bi[n0 + 32 + lanen];
    size_t base = ((((size_t)t * LNUM + l) * 2 + bhalf) * 24) * 32 * 8;
    __bf16* f0 = out + base + ((size_t)(nt0 + 0) * 32 + lane) * 8;
    __bf16* f1 = out + base + ((size_t)(nt0 + 1) * 32 + lane) * 8;
    __bf16* f2 = out + base + ((size_t)(nt0 + 2) * 32 + lane) * 8;
#pragma unroll
    for (int r = 0; r < 8; ++r) {
      f0[r] = (__bf16)(c0[r] + bv0);
      f1[r] = (__bf16)(c1[r] + bv1);
      f2[r] = (__bf16)(c2[r] + bv2);
    }
    return;
  }

#pragma unroll
  for (int r = 0; r < 8; ++r) {
    int m = laneh * 8 + r;
    zbuf[m][n0 + lanen]      = c0[r] + bi[n0 + lanen];
    zbuf[m][n0 + 16 + lanen] = c1[r] + bi[n0 + 16 + lanen];
    zbuf[m][n0 + 32 + lanen] = c2[r] + bi[n0 + 32 + lanen];
  }
  __syncthreads();

  int row = tid >> 4, sub = tid & 15;
  float s = 0.f, s2 = 0.f;
  for (int c = sub; c < CDIM; c += 16) { float x = zbuf[row][c]; s += x; s2 += x * x; }
  red1[row][sub] = s; red2[row][sub] = s2;
  __syncthreads();
  float ts = 0.f, ts2 = 0.f;
#pragma unroll
  for (int j = 0; j < 16; ++j) { ts += red1[row][j]; ts2 += red2[row][j]; }
  float mu = ts / (float)CDIM;
  float inv = rsqrtf(ts2 / (float)CDIM - mu * mu + 1e-5f);

  const float* g  = gamma + l * CDIM;
  const float* be = beta + l * CDIM;
  __bf16* orow = out + (((size_t)t * LNUM + l) * BDIM + (bhalf * 16 + row)) * CDIM;
  for (int c = sub; c < CDIM; c += 16) {
    float x = (zbuf[row][c] - mu) * inv * g[c] + be[c];
    orow[c] = (__bf16)x;
  }
}

// ------------------------- sequential scan -------------------------
__global__ __launch_bounds__(256) void scan_kernel(const __bf16* __restrict__ Wppk,
                                                   const __bf16* __restrict__ scale,
                                                   const __bf16* __restrict__ zfrag,
                                                   const float* __restrict__ gp,
                                                   const float* __restrict__ betap,
                                                   __bf16* __restrict__ h_all) {
  const int bhalf = blockIdx.x;
  const int b0 = bhalf * 16;
  const int tid = threadIdx.x;
  const int wave = tid >> 5, lane = tid & 31;
  const int lanen = lane & 15, laneh = lane >> 4;
  const int n0 = wave * 48;
  const int nt0 = wave * 3;
  const int row = tid >> 4, sub = tid & 15;

  __shared__ float  hbuf[16][CDIM];
  __shared__ __bf16 hb16[16][CDIM];
  __shared__ float  zbuf[16][CDIM];
  __shared__ float  red1[16][16], red2[16][16];

  for (int i = tid; i < 16 * CDIM; i += 256) {
    (&hbuf[0][0])[i] = 0.f;
    (&hb16[0][0])[i] = (__bf16)0.f;
  }
  __syncthreads();

  for (int t = 0; t < TDIM; ++t) {
    for (int l = 0; l < LNUM; ++l) {
      size_t zbase = ((((size_t)t * LNUM + l) * 2 + bhalf) * 24) * 32 * 8;
      const __bf16* z0 = zfrag + zbase + ((size_t)(nt0 + 0) * 32 + lane) * 8;
      const __bf16* z1 = zfrag + zbase + ((size_t)(nt0 + 1) * 32 + lane) * 8;
      const __bf16* z2 = zfrag + zbase + ((size_t)(nt0 + 2) * 32 + lane) * 8;
      const __bf16* Wb = Wppk + (size_t)l * 2 * CDIM * CDIM;  // kt 12..23 = bottom half

      v8f c0, c1, c2;
#pragma unroll
      for (int r = 0; r < 8; ++r) {
        c0[r] = (float)z0[r];
        c1[r] = (float)z1[r];
        c2[r] = (float)z2[r];
      }
      for (int kc = 0; kc < CDIM; kc += 32) {
        int kt = 12 + (kc >> 5);
        v16bf a  = load_a_bf16lds(&hb16[0][0], CDIM, kc, lane);
        v16bf b0 = load_b_packed(Wb, 24, kt, nt0, lane);
        c0 = WMMA_BF16(c0, a, b0);
        v16bf b1 = load_b_packed(Wb, 24, kt, nt0 + 1, lane);
        c1 = WMMA_BF16(c1, a, b1);
        v16bf b2 = load_b_packed(Wb, 24, kt, nt0 + 2, lane);
        c2 = WMMA_BF16(c2, a, b2);
      }
#pragma unroll
      for (int r = 0; r < 8; ++r) {
        int m = laneh * 8 + r;
        zbuf[m][n0 + lanen]      = c0[r];
        zbuf[m][n0 + 16 + lanen] = c1[r];
        zbuf[m][n0 + 32 + lanen] = c2[r];
      }
      __syncthreads();

      float s = 0.f, s2 = 0.f;
      for (int c = sub; c < CDIM; c += 16) { float x = zbuf[row][c]; s += x; s2 += x * x; }
      red1[row][sub] = s; red2[row][sub] = s2;
      __syncthreads();
      float ts = 0.f, ts2 = 0.f;
#pragma unroll
      for (int j = 0; j < 16; ++j) { ts += red1[row][j]; ts2 += red2[row][j]; }
      float mu = ts / (float)CDIM;
      float inv = rsqrtf(ts2 / (float)CDIM - mu * mu + 1e-5f);

      const float* g  = gp + l * CDIM;
      const float* be = betap + l * CDIM;
      const __bf16* sc = scale + (((size_t)t * LNUM + l) * BDIM + b0 + row) * CDIM;
      for (int c = sub; c < CDIM; c += 16) {
        float zn = (zbuf[row][c] - mu) * inv * g[c] + be[c];
        float x = hbuf[row][c] + zn * (float)sc[c];
        float hs = x / (1.f + __expf(-x));   // silu
        hbuf[row][c] = hs;
        hb16[row][c] = (__bf16)hs;
      }
      __syncthreads();
    }
    __bf16* hr = h_all + ((size_t)t * BDIM + b0) * CDIM;
    for (int i = tid; i < 16 * CDIM; i += 256) {
      hr[i] = (&hb16[0][0])[i];
    }
    __syncthreads();
  }
}

// ------------------------- head GEMM -------------------------
__global__ __launch_bounds__(256) void head_kernel(const __bf16* __restrict__ h_all,
                                                   const __bf16* __restrict__ Epk,
                                                   const float* __restrict__ head_b,
                                                   float* __restrict__ out) {
  const int m0 = blockIdx.y * 16;
  const int tid = threadIdx.x;
  const int wave = tid >> 5, lane = tid & 31;
  const int lanen = lane & 15, laneh = lane >> 4;
  const int n0 = blockIdx.x * 256 + wave * 32;
  const int nt0 = n0 >> 4;

  v8f c0 = {}, c1 = {};
  for (int kc = 0; kc < CDIM; kc += 32) {
    int kt = kc >> 5;
    v16bf a  = load_a_bf16(h_all, CDIM, m0, kc, lane);
    v16bf b0 = load_b_packed(Epk, VOCAB / 16, kt, nt0, lane);
    c0 = WMMA_BF16(c0, a, b0);
    v16bf b1 = load_b_packed(Epk, VOCAB / 16, kt, nt0 + 1, lane);
    c1 = WMMA_BF16(c1, a, b1);
  }
  float hb0 = head_b[n0 + lanen];
  float hb1 = head_b[n0 + 16 + lanen];
#pragma unroll
  for (int r = 0; r < 8; ++r) {
    int m = laneh * 8 + r;
    int R = m0 + m;
    int tt = R >> 5, bb = R & 31;
    float* orow = out + ((size_t)bb * TDIM + tt) * VOCAB;
    orow[n0 + lanen]      = c0[r] + hb0;
    orow[n0 + 16 + lanen] = c1[r] + hb1;
  }
}

// ------------------------- launcher -------------------------

extern "C" void kernel_launch(void* const* d_in, const int* in_sizes, int n_in,
                              void* d_out, int out_size, void* d_ws, size_t ws_size,
                              hipStream_t stream) {
  const int*   idx    = (const int*)d_in[0];
  const float* E      = (const float*)d_in[1];
  const float* head_b = (const float*)d_in[2];
  const float* Ws     = (const float*)d_in[3];
  const float* bs     = (const float*)d_in[4];
  const float* gs     = (const float*)d_in[5];
  const float* betas  = (const float*)d_in[6];
  const float* Wp     = (const float*)d_in[7];
  const float* bp     = (const float*)d_in[8];
  const float* gp     = (const float*)d_in[9];
  const float* betap  = (const float*)d_in[10];
  float* out = (float*)d_out;

  char* ws = (char*)d_ws;
  size_t off = 0;
  auto alloc = [&](size_t bytes) -> void* {
    void* p = ws + off;
    off = (off + bytes + 255) & ~((size_t)255);
    return p;
  };
  __bf16* Epk   = (__bf16*)alloc((size_t)VOCAB * CDIM * 2);
  __bf16* Wspk  = (__bf16*)alloc((size_t)LNUM * CDIM * CDIM * 2);
  __bf16* Wppk  = (__bf16*)alloc((size_t)LNUM * 2 * CDIM * CDIM * 2);
  __bf16* emb   = (__bf16*)alloc((size_t)ROWS * CDIM * 2);
  __bf16* scl   = (__bf16*)alloc((size_t)TDIM * LNUM * BDIM * CDIM * 2);
  __bf16* zfrag = (__bf16*)alloc((size_t)TDIM * LNUM * BDIM * CDIM * 2);
  __bf16* h_all = (__bf16*)alloc((size_t)ROWS * CDIM * 2);

  // 1) fused cast+pack of all B operands
  {
    size_t per = (size_t)CDIM * VOCAB;   // B = E^T: K=C, N=VOCAB
    pack_b_T<<<(unsigned)((per + 255) / 256), 256, 0, stream>>>(E, Epk, CDIM, VOCAB);
  }
  {
    int per = CDIM * CDIM;
    dim3 grid((per + 255) / 256, LNUM);
    pack_b<<<grid, 256, 0, stream>>>(Ws, Wspk, CDIM, CDIM);
  }
  {
    int per = 2 * CDIM * CDIM;
    dim3 grid((per + 255) / 256, LNUM);
    pack_b<<<grid, 256, 0, stream>>>(Wp, Wppk, 2 * CDIM, CDIM);
  }
  // 2) embedding gather (time-major rows r = t*B + b)
  {
    int n = ROWS * CDIM;
    gather_emb<<<(n + 255) / 256, 256, 0, stream>>>(idx, E, emb);
  }
  // 3) phase-1 parallel GEMMs
  {
    dim3 grid(ROWS / 16, LNUM);
    gemm_pre<<<grid, 256, 0, stream>>>(emb, Wppk, bp, gs, betas, zfrag,
                                       2 * CDIM * CDIM, 0);   // zpre (fragment layout)
    gemm_pre<<<grid, 256, 0, stream>>>(emb, Wspk, bs, gs, betas, scl,
                                       CDIM * CDIM, 1);       // scale (row-major)
  }
  // 4) sequential scan (2 independent halves of the batch)
  scan_kernel<<<2, 256, 0, stream>>>(Wppk, scl, zfrag, gp, betap, h_all);
  // 5) head GEMM: [8192,384] x [384,32000] + head_b
  {
    dim3 grid(VOCAB / 256, ROWS / 16);
    head_kernel<<<grid, 256, 0, stream>>>(h_all, Epk, head_b, out);
  }
}
